// CrossAttentionModule_30434138259507
// MI455X (gfx1250) — matrix operand order repaired and checked by reading, not compiled, not run on hardware
//
#include <hip/hip_runtime.h>
#include <hip/hip_bf16.h>

// ---------------------------------------------------------------------------
// CrossAttention transformer pair for MI455X (gfx1250), wave32 + WMMA f16.
// B=16, N=576, D=768, H=8, HD=96.  All GEMMs use V_WMMA_F32_16X16X32_F16.
// ---------------------------------------------------------------------------

#define BB   16
#define NN   576
#define DD   768
#define HH   8
#define HDD  96
#define DFF  3072
#define ROWS (BB * NN)            // 9216
#define BHN  (BB * HH)            // 128

typedef __attribute__((ext_vector_type(16))) _Float16 f16x16;
typedef __attribute__((ext_vector_type(8)))  _Float16 f16x8;
typedef __attribute__((ext_vector_type(8)))  float    f32x8;

__device__ __forceinline__ f16x16 cat16(f16x8 lo, f16x8 hi) {
  f16x16 r;
#pragma unroll
  for (int i = 0; i < 8; ++i) { r[i] = lo[i]; r[i + 8] = hi[i]; }
  return r;
}

// A fragment: 16x32 f16, row-major source with leading dim lda (halves).
// lane<16 : row=lane, halves {k0..k0+7, k0+16..k0+23}
// lane>=16: row=lane-16, halves {k0+8..k0+15, k0+24..k0+31}
__device__ __forceinline__ f16x16 load_a(const _Float16* __restrict__ p, int k0) {
  f16x8 lo = *(const f16x8*)(p + k0);
  f16x8 hi = *(const f16x8*)(p + k0 + 16);
  return cat16(lo, hi);
}

// B fragment: 32x16 f16 from transposed storage Bt[col][k] (rows contiguous).
// lane<16 : col=lane,  K {k0..k0+7, k0+16..k0+23};  lane>=16: col=lane-16, K {k0+8..k0+15, k0+24..k0+31}
__device__ __forceinline__ f16x16 load_b(const _Float16* __restrict__ p, int k0) {
  f16x8 lo = *(const f16x8*)(p + k0);
  f16x8 hi = *(const f16x8*)(p + k0 + 16);
  return cat16(lo, hi);
}

__device__ __forceinline__ float gelu_exact(float v) {
  return 0.5f * v * (1.0f + erff(v * 0.70710678118654752f));
}

// ---------------------------------------------------------------------------
// Generic f16 GEMM:  C[M,Nout] = A[M,K] * Bt[Nout,K]^T  (+bias) (+gelu)
// Template: AF x BF 16x16 fragments per wave; MW x NW waves per block.
// Wave tile = (16*AF) x (16*BF); block tile = (16*AF*MW) x (16*BF*NW).
// modes: 0 plain row-major (f16 and/or f32 out)
//        1 Q/K head scatter  -> [B,H,N,HD] f16
//        2 V transposed      -> [B,H,HD,N] f16
//        3 head merge (z=bh) -> [B,N,D]    f16
// ---------------------------------------------------------------------------
template<int AF, int BF, int MW, int NW>
__global__ __launch_bounds__(32 * MW * NW)
void gemm_f16_t(const _Float16* __restrict__ A, const _Float16* __restrict__ Bt,
                const float* __restrict__ bias,
                _Float16* __restrict__ outh, float* __restrict__ outf,
                int M, int Nout, int K, int lda, int ldb, int ldo,
                long long sA, long long sB, long long sO,
                int mode, int gelu)
{
  const int lane = threadIdx.x & 31;
  const int wid  = threadIdx.x >> 5;
  const int z    = blockIdx.z;

  const _Float16* Az = A + (size_t)z * (size_t)sA;
  const _Float16* Bz = Bt + (size_t)z * (size_t)sB;

  const int mrow0 = (blockIdx.x * MW + (wid % MW)) * (16 * AF);
  const int ncol0 = (blockIdx.y * NW + (wid / MW)) * (16 * BF);

  const int as = (lane < 16) ? 0 : 8;
  const int bt = (lane < 16) ? 0 : 8;

  const _Float16* ap[AF];
  const _Float16* bp[BF];
#pragma unroll
  for (int i = 0; i < AF; ++i) {
    int r = mrow0 + i * 16 + (lane & 15); if (r >= M) r = M - 1;
    ap[i] = Az + (size_t)r * lda + as;
  }
#pragma unroll
  for (int j = 0; j < BF; ++j) {
    int c = ncol0 + j * 16 + (lane & 15); if (c >= Nout) c = Nout - 1;
    bp[j] = Bz + (size_t)c * ldb + bt;
  }

  f32x8 acc[AF][BF];
#pragma unroll
  for (int i = 0; i < AF; ++i)
#pragma unroll
    for (int j = 0; j < BF; ++j)
#pragma unroll
      for (int e = 0; e < 8; ++e) acc[i][j][e] = 0.0f;

  for (int k0 = 0; k0 < K; k0 += 32) {
    f16x16 a[AF], b[BF];
#pragma unroll
    for (int i = 0; i < AF; ++i) a[i] = load_a(ap[i], k0);
#pragma unroll
    for (int j = 0; j < BF; ++j) b[j] = load_b(bp[j], k0);
#pragma unroll
    for (int i = 0; i < AF; ++i)
#pragma unroll
      for (int j = 0; j < BF; ++j)
        acc[i][j] = __builtin_amdgcn_wmma_f32_16x16x32_f16(
            false, a[i], false, b[j], (short)0, acc[i][j], false, false);
  }

#pragma unroll
  for (int i = 0; i < AF; ++i) {
    const int rbase = mrow0 + i * 16 + ((lane >> 4) << 3);
#pragma unroll
    for (int j = 0; j < BF; ++j) {
      const int col = ncol0 + j * 16 + (lane & 15);
      if (col >= Nout) continue;
      const float bv = bias ? bias[col] : 0.0f;
#pragma unroll
      for (int r = 0; r < 8; ++r) {
        const int row = rbase + r;
        if (row >= M) break;
        float v = acc[i][j][r] + bv;
        if (gelu) v = gelu_exact(v);
        if (mode == 0) {
          const size_t idx = (size_t)z * (size_t)sO + (size_t)row * ldo + col;
          if (outh) outh[idx] = (_Float16)v;
          if (outf) outf[idx] = v;
        } else if (mode == 1) {
          const int b_ = row / NN, n = row - b_ * NN;
          const int h = col / HDD, hd = col - h * HDD;
          outh[(((size_t)(b_ * HH + h) * NN) + n) * HDD + hd] = (_Float16)v;
        } else if (mode == 2) {
          const int b_ = row / NN, n = row - b_ * NN;
          const int h = col / HDD, hd = col - h * HDD;
          outh[(((size_t)(b_ * HH + h) * HDD) + hd) * NN + n] = (_Float16)v;
        } else { // mode 3: z = b*H + h, row = token, col = hd
          const int b_ = z >> 3, h = z & 7;
          outh[((size_t)(b_ * NN + row)) * DD + h * HDD + col] = (_Float16)v;
        }
      }
    }
  }
}

// ---------------------------------------------------------------------------
// Fused attention scores + softmax:  P = softmax(Q K^T * scale) (f16 out)
// grid (N/16, B*H); block 128 = 4 waves; each wave: 16 q-rows x 144 keys.
// f32 scores staged in LDS (stride 584 to avoid bank conflicts).
// ---------------------------------------------------------------------------
__global__ __launch_bounds__(128)
void scores_softmax(const _Float16* __restrict__ Q, const _Float16* __restrict__ Km,
                    _Float16* __restrict__ P)
{
  __shared__ float S[16 * 584];
  __shared__ float red[16 * 8];
  __shared__ float rowstat[16];
  const int LD = 584;

  const int lane = threadIdx.x & 31;
  const int wid  = threadIdx.x >> 5;
  const int bh   = blockIdx.y;
  const int qrow0 = blockIdx.x * 16;

  const _Float16* Qb = Q  + (size_t)bh * NN * HDD;
  const _Float16* Kb = Km + (size_t)bh * NN * HDD;
  const int kbase = wid * 144;

  f32x8 acc[9];
#pragma unroll
  for (int j = 0; j < 9; ++j)
#pragma unroll
    for (int e = 0; e < 8; ++e) acc[j][e] = 0.0f;

  const int as = (lane < 16) ? 0 : 8;
  const int bt = (lane < 16) ? 0 : 8;
  const _Float16* ap = Qb + (size_t)(qrow0 + (lane & 15)) * HDD + as;

  for (int kd = 0; kd < HDD; kd += 32) {
    f16x16 a = load_a(ap, kd);
#pragma unroll
    for (int j = 0; j < 9; ++j) {
      const _Float16* bp = Kb + (size_t)(kbase + j * 16 + (lane & 15)) * HDD + bt;
      f16x16 b = load_b(bp, kd);
      acc[j] = __builtin_amdgcn_wmma_f32_16x16x32_f16(false, a, false, b, (short)0, acc[j], false, false);
    }
  }

  const float scale = 0.10206207261596575f;  // 96^-0.5
  const int rbase = (lane >> 4) << 3;
#pragma unroll
  for (int j = 0; j < 9; ++j)
#pragma unroll
    for (int r = 0; r < 8; ++r)
      S[(rbase + r) * LD + kbase + j * 16 + (lane & 15)] = acc[j][r] * scale;
  __syncthreads();

  const int row = threadIdx.x >> 3;   // 0..15
  const int seg = threadIdx.x & 7;    // 0..7, 72 cols each
  float m = -1e30f;
  for (int i = 0; i < 72; ++i) m = fmaxf(m, S[row * LD + seg * 72 + i]);
  red[row * 8 + seg] = m;
  __syncthreads();
  if (seg == 0) {
    float mm = red[row * 8];
    for (int i = 1; i < 8; ++i) mm = fmaxf(mm, red[row * 8 + i]);
    rowstat[row] = mm;
  }
  __syncthreads();
  const float rm = rowstat[row];
  float s = 0.f;
  for (int i = 0; i < 72; ++i) {
    float e = __expf(S[row * LD + seg * 72 + i] - rm);
    S[row * LD + seg * 72 + i] = e;
    s += e;
  }
  red[row * 8 + seg] = s;
  __syncthreads();
  if (seg == 0) {
    float ss = 0.f;
    for (int i = 0; i < 8; ++i) ss += red[row * 8 + i];
    rowstat[row] = ss;
  }
  __syncthreads();
  const float inv = 1.0f / rowstat[row];
  _Float16* Pr = P + ((size_t)bh * NN + qrow0 + row) * NN;
  for (int i = 0; i < 72; ++i)
    Pr[seg * 72 + i] = (_Float16)(S[row * LD + seg * 72 + i] * inv);
}

// ---------------------------------------------------------------------------
// out = LayerNorm(x + y) * g + be  (one block per row of 768), optional f16 copy
// ---------------------------------------------------------------------------
__global__ __launch_bounds__(256)
void add_ln(const float* __restrict__ x, const float* __restrict__ y,
            const float* __restrict__ g, const float* __restrict__ be,
            float* __restrict__ out32, _Float16* __restrict__ out16)
{
  __shared__ float red[256];
  const int t = threadIdx.x;
  const size_t base = (size_t)blockIdx.x * DD;
  float v0 = x[base + t]       + y[base + t];
  float v1 = x[base + t + 256] + y[base + t + 256];
  float v2 = x[base + t + 512] + y[base + t + 512];
  red[t] = v0 + v1 + v2;
  __syncthreads();
  for (int o = 128; o > 0; o >>= 1) { if (t < o) red[t] += red[t + o]; __syncthreads(); }
  const float mu = red[0] * (1.0f / DD);
  __syncthreads();
  const float d0 = v0 - mu, d1 = v1 - mu, d2 = v2 - mu;
  red[t] = d0 * d0 + d1 * d1 + d2 * d2;
  __syncthreads();
  for (int o = 128; o > 0; o >>= 1) { if (t < o) red[t] += red[t + o]; __syncthreads(); }
  const float inv = rsqrtf(red[0] * (1.0f / DD) + 1e-5f);
  float o0 = d0 * inv * g[t]       + be[t];
  float o1 = d1 * inv * g[t + 256] + be[t + 256];
  float o2 = d2 * inv * g[t + 512] + be[t + 512];
  out32[base + t] = o0; out32[base + t + 256] = o1; out32[base + t + 512] = o2;
  if (out16) {
    out16[base + t] = (_Float16)o0;
    out16[base + t + 256] = (_Float16)o1;
    out16[base + t + 512] = (_Float16)o2;
  }
}

// ---------------------------------------------------------------------------
// attn_map = 0.5 * (P1 + P2^T)   over [B*H, N, N]
// ---------------------------------------------------------------------------
__global__ __launch_bounds__(256)
void combine_attn(const _Float16* __restrict__ P1, const _Float16* __restrict__ P2,
                  float* __restrict__ out)
{
  const size_t idx = (size_t)blockIdx.x * 256 + threadIdx.x;
  const int k = (int)(idx % NN);
  const size_t t = idx / NN;
  const int q = (int)(t % NN);
  const size_t bh = t / NN;
  const float a = (float)P1[idx];
  const float b = (float)P2[(bh * NN + k) * NN + q];
  out[idx] = 0.5f * (a + b);
}

// W [Kdim, Ndim] f32  ->  Wt [Ndim, Kdim] f16
__global__ __launch_bounds__(256)
void transpose_f16(const float* __restrict__ W, _Float16* __restrict__ Wt,
                   int Kdim, int Ndim)
{
  const size_t i = (size_t)blockIdx.x * 256 + threadIdx.x;
  if (i >= (size_t)Kdim * Ndim) return;
  const int k = (int)(i / Ndim);
  const int n = (int)(i % Ndim);
  Wt[(size_t)n * Kdim + k] = (_Float16)W[i];
}

__global__ __launch_bounds__(256)
void to_f16(const float* __restrict__ src, _Float16* __restrict__ dst, size_t n)
{
  const size_t i = (size_t)blockIdx.x * 256 + threadIdx.x;
  if (i < n) dst[i] = (_Float16)src[i];
}

// ---------------------------------------------------------------------------
extern "C" void kernel_launch(void* const* d_in, const int* in_sizes, int n_in,
                              void* d_out, int out_size, void* d_ws, size_t ws_size,
                              hipStream_t stream)
{
  (void)in_sizes; (void)n_in; (void)out_size; (void)ws_size;
  const float* x1  = (const float*)d_in[0];
  const float* x2  = (const float*)d_in[1];
  const float* Wq  = (const float*)d_in[2];
  const float* bq  = (const float*)d_in[3];
  const float* Wk  = (const float*)d_in[4];
  const float* bk  = (const float*)d_in[5];
  const float* Wv  = (const float*)d_in[6];
  const float* bv  = (const float*)d_in[7];
  const float* Wo  = (const float*)d_in[8];
  const float* bo  = (const float*)d_in[9];
  const float* g1  = (const float*)d_in[10];
  const float* be1 = (const float*)d_in[11];
  const float* g2  = (const float*)d_in[12];
  const float* be2 = (const float*)d_in[13];
  const float* W1  = (const float*)d_in[14];
  const float* bf1 = (const float*)d_in[15];
  const float* W2  = (const float*)d_in[16];
  const float* bf2 = (const float*)d_in[17];

  float* out  = (float*)d_out;
  float* out1 = out;
  float* out2 = out + (size_t)ROWS * DD;
  float* attn = out + 2 * (size_t)ROWS * DD;

  // workspace carve-up (256B aligned)
  size_t off = 0;
  auto alloc = [&](size_t bytes) -> void* {
    void* p = (char*)d_ws + off;
    off += (bytes + 255) & ~(size_t)255;
    return p;
  };
  const size_t XH   = (size_t)ROWS * DD * 2;       // 14.2 MB
  const size_t WSQ  = (size_t)DD * DD * 2;         // 1.2 MB
  const size_t WFF  = (size_t)DD * DFF * 2;        // 4.7 MB
  const size_t QKV  = (size_t)BHN * NN * HDD * 2;  // 14.2 MB
  const size_t PSZ  = (size_t)BHN * NN * NN * 2;   // 84.9 MB
  _Float16* x1h = (_Float16*)alloc(XH);
  _Float16* x2h = (_Float16*)alloc(XH);
  _Float16* Wqt = (_Float16*)alloc(WSQ);
  _Float16* Wkt = (_Float16*)alloc(WSQ);
  _Float16* Wvt = (_Float16*)alloc(WSQ);
  _Float16* Wot = (_Float16*)alloc(WSQ);
  _Float16* W1t = (_Float16*)alloc(WFF);
  _Float16* W2t = (_Float16*)alloc(WFF);
  _Float16* Qh[2] = {(_Float16*)alloc(QKV), (_Float16*)alloc(QKV)};
  _Float16* Kh[2] = {(_Float16*)alloc(QKV), (_Float16*)alloc(QKV)};
  _Float16* Vt[2] = {(_Float16*)alloc(QKV), (_Float16*)alloc(QKV)};
  _Float16* P[2]  = {(_Float16*)alloc(PSZ), (_Float16*)alloc(PSZ)};
  _Float16* Oh[2] = {(_Float16*)alloc(XH), (_Float16*)alloc(XH)};
  float*    tmp32 = (float*)alloc((size_t)ROWS * DD * 4);
  float*    hf32  = (float*)alloc((size_t)ROWS * DD * 4);
  _Float16* hh16  = (_Float16*)alloc(XH);
  _Float16* G     = (_Float16*)alloc((size_t)ROWS * DFF * 2);

  // --- stage 0: convert inputs and transpose weights to f16 [out,in] ---
  transpose_f16<<<2304, 256, 0, stream>>>(Wq, Wqt, DD, DD);
  transpose_f16<<<2304, 256, 0, stream>>>(Wk, Wkt, DD, DD);
  transpose_f16<<<2304, 256, 0, stream>>>(Wv, Wvt, DD, DD);
  transpose_f16<<<2304, 256, 0, stream>>>(Wo, Wot, DD, DD);
  transpose_f16<<<9216, 256, 0, stream>>>(W1, W1t, DD, DFF);
  transpose_f16<<<9216, 256, 0, stream>>>(W2, W2t, DFF, DD);
  to_f16<<<27648, 256, 0, stream>>>(x1, x1h, (size_t)ROWS * DD);
  to_f16<<<27648, 256, 0, stream>>>(x2, x2h, (size_t)ROWS * DD);

  // big GEMM config: wave 32x64, block 64x256, 256 threads
  #define GEMM_BIG gemm_f16_t<2, 4, 2, 4>
  // P*V config: wave 32x32, block 64x96, 192 threads (no tile waste at N=96)
  #define GEMM_PV  gemm_f16_t<2, 2, 2, 3>

  // --- stage 1: QKV projections, attention scores+softmax, P*V ---
  for (int dir = 0; dir < 2; ++dir) {
    const _Float16* xq  = dir ? x2h : x1h;
    const _Float16* xkv = dir ? x1h : x2h;
    GEMM_BIG<<<dim3(144, 3, 1), 256, 0, stream>>>(xq,  Wqt, bq, Qh[dir], nullptr,
        ROWS, DD, DD, DD, DD, 0, 0, 0, 0, /*mode*/1, 0);
    GEMM_BIG<<<dim3(144, 3, 1), 256, 0, stream>>>(xkv, Wkt, bk, Kh[dir], nullptr,
        ROWS, DD, DD, DD, DD, 0, 0, 0, 0, /*mode*/1, 0);
    GEMM_BIG<<<dim3(144, 3, 1), 256, 0, stream>>>(xkv, Wvt, bv, Vt[dir], nullptr,
        ROWS, DD, DD, DD, DD, 0, 0, 0, 0, /*mode*/2, 0);
    scores_softmax<<<dim3(36, BHN, 1), 128, 0, stream>>>(Qh[dir], Kh[dir], P[dir]);
    GEMM_PV<<<dim3(9, 1, BHN), 192, 0, stream>>>(P[dir], Vt[dir], nullptr, Oh[dir], nullptr,
        NN, HDD, NN, NN, NN, 0, (long long)NN * NN, (long long)HDD * NN, 0, /*mode*/3, 0);
  }

  // --- stage 2: O-proj + LN1 + FFN + LN2 per direction (shared scratch) ---
  for (int dir = 0; dir < 2; ++dir) {
    const float* xres = dir ? x2 : x1;
    GEMM_BIG<<<dim3(144, 3, 1), 256, 0, stream>>>(Oh[dir], Wot, bo, nullptr, tmp32,
        ROWS, DD, DD, DD, DD, DD, 0, 0, 0, /*mode*/0, 0);
    add_ln<<<ROWS, 256, 0, stream>>>(xres, tmp32, g1, be1, hf32, hh16);
    GEMM_BIG<<<dim3(144, 12, 1), 256, 0, stream>>>(hh16, W1t, bf1, G, nullptr,
        ROWS, DFF, DD, DD, DD, DFF, 0, 0, 0, /*mode*/0, /*gelu*/1);
    GEMM_BIG<<<dim3(144, 3, 1), 256, 0, stream>>>(G, W2t, bf2, nullptr, tmp32,
        ROWS, DD, DFF, DFF, DFF, DD, 0, 0, 0, /*mode*/0, 0);
    add_ln<<<ROWS, 256, 0, stream>>>(hf32, tmp32, g2, be2, dir ? out2 : out1, nullptr);
  }

  // --- stage 3: attention map ---
  combine_attn<<<(BHN * (size_t)NN * NN) / 256, 256, 0, stream>>>(P[0], P[1], attn);
}
